// CrossAttention_72636486910511
// MI455X (gfx1250) — compile-verified, matrix operand order checked by reading, not run on hardware
//
#include <hip/hip_runtime.h>
#include <hip/hip_bf16.h>
#include <math.h>

// ---------------------------------------------------------------- constants
constexpr int B_ = 4;
constexpr int N_ = 2048;
constexpr int M_ = 2048;
constexpr int F_ = 512;   // features == MID
constexpr int H_ = 8;
constexpr int D_ = 64;

typedef __attribute__((ext_vector_type(16))) _Float16 v16h;
typedef __attribute__((ext_vector_type(8)))  _Float16 v8h;
typedef __attribute__((ext_vector_type(8)))  float    v8f;

// ---------------------------------------------------------------- WMMA core
__device__ __forceinline__ v8f wmma_f16(v16h a, v16h b, v8f c) {
    // D = A(16x32 f16) x B(32x16 f16) + C(16x16 f32)
    return __builtin_amdgcn_wmma_f32_16x16x32_f16(
        /*neg_a=*/false, a, /*neg_b=*/false, b,
        /*c_mod=*/(short)0, c, /*reuse_a=*/false, /*reuse_b=*/false);
}

// Load a 16x32 f16 A-fragment (also used for B-fragments from a transposed
// matrix: B column n == Bt row n). ISA layout (CDNA5 ISA):
//   lane r (0-15) holds row r ; lanes 16-31 hold same rows, other K half.
//   half j (0..15) -> K = k0 + j + 8*((j>=8) + (lane>=16))
// => two contiguous 16-byte loads per lane.
__device__ __forceinline__ v16h load_frag16(const _Float16* __restrict__ base,
                                            int ld, int r0, int k0) {
    int lane = threadIdx.x & 31;
    int r  = lane & 15;
    int lh = lane >> 4;
    const _Float16* p = base + (size_t)(r0 + r) * (size_t)ld + k0;
    v8h c0 = *(const v8h*)(p + 8 * lh);        // j = 0..7
    v8h c1 = *(const v8h*)(p + 16 + 8 * lh);   // j = 8..15
    v16h out;
#pragma unroll
    for (int j = 0; j < 8; ++j) { out[j] = c0[j]; out[j + 8] = c1[j]; }
    return out;
}

// ---------------------------------------------------------------- LayerNorm
// One block per row of 512 fp32; emits f16.
__global__ void __launch_bounds__(256)
ln_cast(const float* __restrict__ in, const float* __restrict__ w,
        const float* __restrict__ bv, _Float16* __restrict__ out) {
    __shared__ float s1[256], s2[256];
    int tid = threadIdx.x;
    const float* p = in + (size_t)blockIdx.x * F_;
    float a = p[tid], b = p[tid + 256];
    s1[tid] = a + b;
    s2[tid] = a * a + b * b;
    __syncthreads();
    for (int off = 128; off > 0; off >>= 1) {
        if (tid < off) { s1[tid] += s1[tid + off]; s2[tid] += s2[tid + off]; }
        __syncthreads();
    }
    float mean = s1[0] * (1.f / F_);
    float var  = s2[0] * (1.f / F_) - mean * mean;
    float rstd = rsqrtf(var + 1e-5f);
    _Float16* q = out + (size_t)blockIdx.x * F_;
    q[tid]       = (_Float16)((a - mean) * rstd * w[tid] + bv[tid]);
    q[tid + 256] = (_Float16)((b - mean) * rstd * w[tid + 256] + bv[tid + 256]);
}

// ---------------------------------------------------------------- W -> Wt f16
// out[n*512 + k] = (f16) in[k*512 + n]   (512x512)
__global__ void __launch_bounds__(256)
transpose_cast(const float* __restrict__ in, _Float16* __restrict__ out) {
    int id = blockIdx.x * 256 + threadIdx.x;
    int k = id >> 9, n = id & 511;
    out[(size_t)n * F_ + k] = (_Float16)in[id];
}

// ---------------------------------------------------------------- GEMM 512-K
// C[rows,512] = A[rows,512] @ W, with W given transposed (Bt[n][k] = W[k][n]).
// One wave computes one 16x16 tile; 8 waves/block cover 128 cols.
// MODE 0: f16 row-major out            (Q, K projections)
// MODE 1: f16 out in [b,h][d][m] layout (V projection -> contiguous PV B-frags)
// MODE 2: f32 out + bias               (final output projection)
template <int MODE>
__global__ void __launch_bounds__(256)
gemm512(const _Float16* __restrict__ A, const _Float16* __restrict__ Bt,
        void* __restrict__ outp, const float* __restrict__ bias) {
    int wave = threadIdx.x >> 5;
    int lane = threadIdx.x & 31;
    int col0 = (blockIdx.x * 8 + wave) * 16;
    int row0 = blockIdx.y * 16;

    v8f acc = {};
#pragma unroll 4
    for (int k0 = 0; k0 < F_; k0 += 32)
        acc = wmma_f16(load_frag16(A, F_, row0, k0),
                       load_frag16(Bt, F_, col0, k0), acc);

    int m_lo = 8 * (lane >> 4);   // C/D layout: vgpr i -> row i (+8 hi lanes)
    int c    = lane & 15;         // N = lane within 16-group
    if (MODE == 0) {
        _Float16* out = (_Float16*)outp;
#pragma unroll
        for (int i = 0; i < 8; ++i)
            out[(size_t)(row0 + m_lo + i) * F_ + col0 + c] = (_Float16)acc[i];
    } else if (MODE == 1) {
        _Float16* out = (_Float16*)outp;
        int col = col0 + c, h = col >> 6, d = col & 63;
#pragma unroll
        for (int i = 0; i < 8; ++i) {
            int row = row0 + m_lo + i;
            int b = row >> 11, m = row & (M_ - 1);
            out[(size_t)((b * H_ + h) * D_ + d) * M_ + m] = (_Float16)acc[i];
        }
    } else {
        float* out = (float*)outp;
        float bb = bias[col0 + c];
#pragma unroll
        for (int i = 0; i < 8; ++i)
            out[(size_t)(row0 + m_lo + i) * F_ + col0 + c] = acc[i] + bb;
    }
}

// ---------------------------------------------------------------- attention
// Flash attention with TRANSPOSED scores: St = K @ Qt (same fragment loads as
// Q @ Kt, operands swapped). In St's C/D layout lane = query, vgpr/lanegroup
// = key, which IS the A-fragment layout required by P@V:
//   A-frag lane l (query l&15, g=l>>4), half j -> key = j + 8*((j>=8)+g)
//   St tile (keys base+0..15) D-layout: lane l, vgpr i -> key = i + 8g
// so pa[j] = exp(t0[j]), pa[j+8] = exp(t1[j]) with NO cross-lane movement.
// Softmax stats are one scalar per lane (its query): single shfl_xor(16) to
// combine the two key-halves, 8 shfl broadcasts to distribute alpha/1/l to
// O rows. 2 query tiles per wave so each K/V fragment feeds 2 WMMAs.
// grid = (B*H, N/256); 8 waves/block; wave = 32 query rows.
// Q [B*N,512] f16 ; Km [M,512] f16 (batch-shared) ; Vt [B*H][64][M] f16
// O [B*N,512] f16
__global__ void __launch_bounds__(256)
attn_kernel(const _Float16* __restrict__ Q, const _Float16* __restrict__ Km,
            const _Float16* __restrict__ Vt, _Float16* __restrict__ O) {
    int wave = threadIdx.x >> 5;
    int lane = threadIdx.x & 31;
    int bh = blockIdx.x;                 // b*H + h
    int b = bh >> 3, h = bh & 7;
    int r0 = blockIdx.y * 256 + wave * 32;
    int qrow0 = b * N_ + r0;
    int hD = h * D_;
    int m_lo = 8 * (lane >> 4);
    int cc   = lane & 15;

    // Q fragments (B operands of St): [tile][d-half]
    v16h aq[2][2];
#pragma unroll
    for (int t = 0; t < 2; ++t) {
        aq[t][0] = load_frag16(Q, F_, qrow0 + 16 * t, hD);
        aq[t][1] = load_frag16(Q, F_, qrow0 + 16 * t, hD + 32);
    }
    const _Float16* Vbh = Vt + (size_t)bh * D_ * M_;

    float mrun[2] = {-INFINITY, -INFINITY};
    float lrun[2] = {0.f, 0.f};
    v8f o[2][4] = {};

    for (int kb = 0; kb < M_; kb += 32) {
        // K fragments (A operands of St) and V fragments: shared by both tiles
        v16h k00 = load_frag16(Km, F_, kb,      hD);
        v16h k01 = load_frag16(Km, F_, kb,      hD + 32);
        v16h k10 = load_frag16(Km, F_, kb + 16, hD);
        v16h k11 = load_frag16(Km, F_, kb + 16, hD + 32);
        v16h vf[4];
#pragma unroll
        for (int n = 0; n < 4; ++n) vf[n] = load_frag16(Vbh, M_, 16 * n, kb);

#pragma unroll
        for (int t = 0; t < 2; ++t) {
            v8f z = {};
            // St tiles: rows = keys (kb.. / kb+16..), cols = queries
            v8f t0 = wmma_f16(k00, aq[t][0], z);
            t0     = wmma_f16(k01, aq[t][1], t0);
            v8f t1 = wmma_f16(k10, aq[t][0], z);
            t1     = wmma_f16(k11, aq[t][1], t1);

            // per-lane softmax for this lane's query (16 keys in-register,
            // the other 16 keys of the chunk live in lane^16)
            float cmax = -INFINITY;
#pragma unroll
            for (int i = 0; i < 8; ++i) {
                t0[i] *= 0.125f;  // D^-0.5
                t1[i] *= 0.125f;
                cmax = fmaxf(cmax, fmaxf(t0[i], t1[i]));
            }
            cmax = fmaxf(cmax, __shfl_xor(cmax, 16, 32));
            float mn    = fmaxf(mrun[t], cmax);
            float alpha = __expf(mrun[t] - mn);
            mrun[t] = mn;

            float rsum = 0.f;
            v16h pa;   // A-fragment of P, built in place (see header comment)
#pragma unroll
            for (int i = 0; i < 8; ++i) {
                float e0 = __expf(t0[i] - mn);
                float e1 = __expf(t1[i] - mn);
                rsum += e0 + e1;
                pa[i]     = (_Float16)e0;
                pa[i + 8] = (_Float16)e1;
            }
            rsum += __shfl_xor(rsum, 16, 32);
            lrun[t] = lrun[t] * alpha + rsum;

            // rescale O rows: row i+m_lo's alpha lives in lane (i+m_lo)
#pragma unroll
            for (int i = 0; i < 8; ++i) {
                float ar = __shfl(alpha, m_lo + i, 32);
                o[t][0][i] *= ar; o[t][1][i] *= ar;
                o[t][2][i] *= ar; o[t][3][i] *= ar;
            }

            // O += P(16x32) @ V(32x64)
#pragma unroll
            for (int n = 0; n < 4; ++n)
                o[t][n] = wmma_f16(pa, vf[n], o[t][n]);
        }
    }

    // normalize (1/l broadcast per O row) and store f16 [B*N, 512]
#pragma unroll
    for (int t = 0; t < 2; ++t) {
        float invl = 1.f / lrun[t];
#pragma unroll
        for (int i = 0; i < 8; ++i) {
            float ir = __shfl(invl, m_lo + i, 32);
            size_t ro = (size_t)(qrow0 + 16 * t + m_lo + i) * F_ + hD + cc;
            O[ro]      = (_Float16)(o[t][0][i] * ir);
            O[ro + 16] = (_Float16)(o[t][1][i] * ir);
            O[ro + 32] = (_Float16)(o[t][2][i] * ir);
            O[ro + 48] = (_Float16)(o[t][3][i] * ir);
        }
    }
}

// ---------------------------------------------------------------- launcher
extern "C" void kernel_launch(void* const* d_in, const int* in_sizes, int n_in,
                              void* d_out, int out_size, void* d_ws, size_t ws_size,
                              hipStream_t stream) {
    const float* x       = (const float*)d_in[0];   // [B,N,F]
    const float* context = (const float*)d_in[1];   // [B,M,F]
    const float* pos_emb = (const float*)d_in[2];   // [M,F]
    const float* ln_w    = (const float*)d_in[3];
    const float* ln_b    = (const float*)d_in[4];
    const float* lnc_w   = (const float*)d_in[5];
    const float* lnc_b   = (const float*)d_in[6];
    const float* Wq      = (const float*)d_in[7];   // [F,MID]
    const float* Wk      = (const float*)d_in[8];
    const float* Wv      = (const float*)d_in[9];
    const float* Wout    = (const float*)d_in[10];  // [MID,F]
    const float* bout    = (const float*)d_in[11];

    char* ws = (char*)d_ws;
    size_t off = 0;
    auto alloc = [&](size_t bytes) -> void* {
        void* p = ws + off;
        off += (bytes + 255) & ~(size_t)255;
        return p;
    };
    _Float16* xn    = (_Float16*)alloc((size_t)B_ * N_ * F_ * 2);
    _Float16* cn    = (_Float16*)alloc((size_t)B_ * M_ * F_ * 2);
    _Float16* kn    = (_Float16*)alloc((size_t)M_ * F_ * 2);
    _Float16* WqT   = (_Float16*)alloc((size_t)F_ * F_ * 2);
    _Float16* WkT   = (_Float16*)alloc((size_t)F_ * F_ * 2);
    _Float16* WvT   = (_Float16*)alloc((size_t)F_ * F_ * 2);
    _Float16* WoutT = (_Float16*)alloc((size_t)F_ * F_ * 2);
    _Float16* Qf    = (_Float16*)alloc((size_t)B_ * N_ * F_ * 2);
    _Float16* Kf    = (_Float16*)alloc((size_t)M_ * F_ * 2);
    _Float16* Vtf   = (_Float16*)alloc((size_t)B_ * M_ * F_ * 2);
    _Float16* Of    = (_Float16*)alloc((size_t)B_ * N_ * F_ * 2);

    // 1) layernorm + f16 cast (keys are batch-invariant: done once on pos_emb)
    ln_cast<<<B_ * N_, 256, 0, stream>>>(x, ln_w, ln_b, xn);
    ln_cast<<<M_,      256, 0, stream>>>(pos_emb, ln_w, ln_b, kn);
    ln_cast<<<B_ * M_, 256, 0, stream>>>(context, lnc_w, lnc_b, cn);

    // 2) weights -> transposed f16 (contiguous WMMA B-fragments)
    transpose_cast<<<1024, 256, 0, stream>>>(Wq, WqT);
    transpose_cast<<<1024, 256, 0, stream>>>(Wk, WkT);
    transpose_cast<<<1024, 256, 0, stream>>>(Wv, WvT);
    transpose_cast<<<1024, 256, 0, stream>>>(Wout, WoutT);

    // 3) projections
    gemm512<0><<<dim3(4, (B_ * N_) / 16), 256, 0, stream>>>(xn, WqT, Qf, nullptr);
    gemm512<0><<<dim3(4, M_ / 16),        256, 0, stream>>>(kn, WkT, Kf, nullptr);
    gemm512<1><<<dim3(4, (B_ * M_) / 16), 256, 0, stream>>>(cn, WvT, Vtf, nullptr);

    // 4) fused flash attention (transposed-score formulation)
    attn_kernel<<<dim3(B_ * H_, N_ / 256), 256, 0, stream>>>(Qf, Kf, Vtf, Of);

    // 5) output projection + bias (fp32 out)
    gemm512<2><<<dim3(4, (B_ * N_) / 16), 256, 0, stream>>>(Of, WoutT, (float*)d_out, bout);
}
